// GatedDeltaCore_62947040690677
// MI455X (gfx1250) — matrix operand order repaired and checked by reading, not compiled, not run on hardware
//
#include <hip/hip_runtime.h>
#include <hip/hip_bf16.h>
#include <math.h>
#include <stdint.h>

typedef __bf16 bf16;
typedef __attribute__((ext_vector_type(16))) __bf16 v16bf;
typedef __attribute__((ext_vector_type(8)))  float  v8f;

#define DIM 2048
#define HD  128
#define NH  16
#define CHK 128
#define BB  2
#define TT  4096
#define NCH 32          // TT / CHK
#define BT  (BB * TT)   // 8192

// ---------------------------------------------------------------- WMMA utils

__device__ __forceinline__ v8f wmma_bf16(v16bf a, v16bf b, v8f c) {
  return __builtin_amdgcn_wmma_f32_16x16x32_bf16(false, a, false, b,
                                                 (short)0, c, false, false);
}

// A fragment 16x32 (bf16): source base[row*ld + k], row-major.
__device__ __forceinline__ v16bf load_frag_a(const bf16* base, int ld,
                                             int row0, int k0, int lane) {
  int r = row0 + (lane & 15);
  int kadd = (lane & 16) ? 16 : 0;
  const bf16* p = base + (size_t)r * ld + k0 + kadd;
  v16bf f;
#pragma unroll
  for (int j = 0; j < 16; ++j) {
    int k = j;
    f[j] = p[k];
  }
  return f;
}

// B fragment 32x16 (bf16) from transposed storage base[col*ld + k].
__device__ __forceinline__ v16bf load_frag_b(const bf16* base, int ld,
                                             int col0, int k0, int lane) {
  int c = col0 + (lane & 15);
  int kadd = (lane & 16) ? 16 : 0;
  const bf16* p = base + (size_t)c * ld + k0 + kadd;
  v16bf f;
#pragma unroll
  for (int j = 0; j < 16; ++j) f[j] = p[j];
  return f;
}

__device__ __forceinline__ void cd_coord(int lane, int v, int& row, int& col) {
  col = lane & 15;
  row = (lane & 16) ? (v + 8) : v;
}

// ------------------------------------------------- CDNA5 async global->LDS
__device__ __forceinline__ void async_load_lds_b128(void* lds_ptr,
                                                    const void* gptr) {
  uint32_t lds_addr = (uint32_t)(uintptr_t)lds_ptr;
  asm volatile("global_load_async_to_lds_b128 %0, %1, off"
               :: "v"(lds_addr), "v"(gptr)
               : "memory");
}
__device__ __forceinline__ void wait_asynccnt0() {
  asm volatile("s_wait_asynccnt 0" ::: "memory");
}

// ---------------------------------------------------------------- converters

__global__ void cvt_f32_bf16(const float* __restrict__ src,
                             bf16* __restrict__ dst, long long n) {
  long long i = (long long)blockIdx.x * blockDim.x + threadIdx.x;
  long long stride = (long long)gridDim.x * blockDim.x;
  for (; i < n; i += stride) dst[i] = (bf16)src[i];
}

// W [k][n] f32 -> Wt [n][k] bf16 (tiled transpose so GEMM B staging is a
// contiguous row copy eligible for async DMA)
__global__ void cvt_transpose_w(const float* __restrict__ src,
                                bf16* __restrict__ dst) {
  __shared__ float tile[32][33];
  const int k0 = blockIdx.x * 32, n0 = blockIdx.y * 32;
  const int tx = threadIdx.x, ty = threadIdx.y;  // 32 x 8
  for (int r = ty; r < 32; r += 8)
    tile[r][tx] = src[(size_t)(k0 + r) * DIM + n0 + tx];
  __syncthreads();
  for (int r = ty; r < 32; r += 8)
    dst[(size_t)(n0 + r) * DIM + k0 + tx] = (bf16)tile[tx][r];
}

// ---------------------------------------------------------------- QKV GEMM
// O = Xb @ WbT^T + bias ; M=8192, N=2048, K=2048, blockIdx.z selects q/k/v.
// Double-buffered LDS; both tiles staged via async global->LDS DMA; bf16 out.
__global__ __launch_bounds__(256) void gemm_qkv(
    const bf16* __restrict__ xb, const bf16* __restrict__ wbt,
    const float* __restrict__ bq, const float* __restrict__ bk,
    const float* __restrict__ bv,
    bf16* __restrict__ qb, bf16* __restrict__ kb, bf16* __restrict__ vb) {
  __shared__ alignas(16) bf16 As[2][128 * 32];   // [row m][k]
  __shared__ alignas(16) bf16 Bs[2][128 * 32];   // [row n][k]
  const int which = blockIdx.z;
  const bf16* Wt = wbt + (size_t)which * DIM * DIM;
  const float* bias = (which == 0) ? bq : (which == 1 ? bk : bv);
  bf16* out = (which == 0) ? qb : (which == 1 ? kb : vb);
  const int n0 = blockIdx.x * 128, m0 = blockIdx.y * 128;
  const int tid = threadIdx.x, lane = tid & 31, wave = tid >> 5;
  const int wm = (wave >> 1) * 32, wn = (wave & 1) * 64;

  auto stage = [&](int buf, int kt) {
#pragma unroll
    for (int u = 0; u < 2; ++u) {
      int i = tid + u * 256;
      int row = i >> 2, part = i & 3;
      async_load_lds_b128(&As[buf][row * 32 + part * 8],
                          &xb[(size_t)(m0 + row) * DIM + kt + part * 8]);
      async_load_lds_b128(&Bs[buf][row * 32 + part * 8],
                          &Wt[(size_t)(n0 + row) * DIM + kt + part * 8]);
    }
  };

  v8f acc[2][4];
#pragma unroll
  for (int i = 0; i < 2; ++i)
#pragma unroll
    for (int j = 0; j < 4; ++j) acc[i][j] = {};

  stage(0, 0);
  wait_asynccnt0();
  __syncthreads();

  int buf = 0;
  for (int kt = 0; kt < DIM; kt += 32) {
    const int nxt = kt + 32;
    if (nxt < DIM) stage(buf ^ 1, nxt);  // DMA next tile during compute
    v16bf a0 = load_frag_a(&As[buf][0], 32, wm, 0, lane);
    v16bf a1 = load_frag_a(&As[buf][0], 32, wm + 16, 0, lane);
#pragma unroll
    for (int ni = 0; ni < 4; ++ni) {
      v16bf bfr = load_frag_b(&Bs[buf][0], 32, wn + ni * 16, 0, lane);
      acc[0][ni] = wmma_bf16(a0, bfr, acc[0][ni]);
      acc[1][ni] = wmma_bf16(a1, bfr, acc[1][ni]);
    }
    if (nxt < DIM) wait_asynccnt0();
    __syncthreads();
    buf ^= 1;
  }
#pragma unroll
  for (int mi = 0; mi < 2; ++mi)
#pragma unroll
    for (int ni = 0; ni < 4; ++ni)
#pragma unroll
      for (int v = 0; v < 8; ++v) {
        int r, c;
        cd_coord(lane, v, r, c);
        int gm = m0 + wm + mi * 16 + r;
        int gn = n0 + wn + ni * 16 + c;
        out[(size_t)gm * DIM + gn] = (bf16)(acc[mi][ni][v] + bias[gn]);
      }
}

// ---------------------------------------------------------------- gates
__global__ __launch_bounds__(256) void proj_gate(
    const float* __restrict__ xf, const float* __restrict__ Wbeta,
    const float* __restrict__ bbeta, const float* __restrict__ Walpha,
    const float* __restrict__ balpha, float* __restrict__ betaw,
    float* __restrict__ law, float* __restrict__ alpha_out) {
  const int token = blockIdx.x;
  const int tid = threadIdx.x, lane = tid & 31, wave = tid >> 5;
  const float* xrow = xf + (size_t)token * DIM;
#pragma unroll
  for (int sub = 0; sub < 4; ++sub) {
    int o = wave * 4 + sub;
    int col = o & 15;
    bool isAlpha = (o >= 16);
    const float* W = isAlpha ? Walpha : Wbeta;
    float s = 0.f;
    for (int k = lane; k < DIM; k += 32) s += xrow[k] * W[(size_t)k * NH + col];
#pragma unroll
    for (int off = 16; off > 0; off >>= 1) s += __shfl_xor(s, off, 32);
    if (lane == 0) {
      if (isAlpha) {
        float z = s + balpha[col];
        float a = 1.f / (1.f + expf(-z));
        alpha_out[(size_t)token * NH + col] = a;
        law[(size_t)token * NH + col] = logf(a + 1e-6f);
      } else {
        float z = s + bbeta[col];
        betaw[(size_t)token * NH + col] = (z > 20.f) ? z : log1pf(expf(z));
      }
    }
  }
}

// ---------------------------------------------------------------- chunk-local
__global__ __launch_bounds__(256) void chunk_local(
    const bf16* __restrict__ qb, const bf16* __restrict__ kb,
    const bf16* __restrict__ vb, const float* __restrict__ betaw,
    const float* __restrict__ law, float* __restrict__ ylocal,
    float* __restrict__ deltaw, float* __restrict__ cdw,
    float* __restrict__ tdw, float* __restrict__ sacc) {
  __shared__ alignas(16) bf16 qs[CHK * HD];   // async-DMA'd; reused for attn
  __shared__ alignas(16) bf16 ks[CHK * HD];   // normalized k, [c][s]
  __shared__ alignas(16) bf16 vt[HD * CHK];   // v transposed: vt[d][c]
  __shared__ alignas(16) bf16 kwt[HD * CHK];  // kw transposed: kwt[e][c]
  __shared__ float beta_s[CHK], cla[CHK], lab[CHK], rowsq[CHK];

  const int h = blockIdx.x, n = blockIdx.y, b = blockIdx.z;
  const int tid = threadIdx.x, lane = tid & 31, wave = tid >> 5;
  const int wm = (wave >> 1) * 32, wn = (wave & 1) * 64;
  const size_t tok0 = (size_t)b * TT + (size_t)n * CHK;
  const size_t bnh = (size_t)(b * NCH + n) * NH + h;

  // q tile: straight bf16 copy -> async DMA (overlaps with k/v staging below)
  for (int i = tid; i < (CHK * HD) / 8; i += 256) {
    int c = i >> 4, part = i & 15;
    async_load_lds_b128(&qs[c * HD + part * 8],
                        &qb[(tok0 + c) * DIM + (size_t)h * HD + part * 8]);
  }
  if (tid < CHK) {
    const int c = tid;
    const size_t rowoff = (tok0 + c) * DIM + (size_t)h * HD;  // 16B aligned
    const uint4* kr = reinterpret_cast<const uint4*>(kb + rowoff);
    const uint4* vr = reinterpret_cast<const uint4*>(vb + rowoff);
    float ss = 0.f;
#pragma unroll
    for (int s8 = 0; s8 < HD / 8; ++s8) {
      uint4 t = kr[s8];
      const bf16* e = reinterpret_cast<const bf16*>(&t);
#pragma unroll
      for (int u = 0; u < 8; ++u) { float kv = (float)e[u]; ss += kv * kv; }
    }
    float rn = 1.f / fmaxf(sqrtf(ss), 1e-12f);
#pragma unroll
    for (int s8 = 0; s8 < HD / 8; ++s8) {
      uint4 tk = kr[s8], tv = vr[s8];
      const bf16* ek = reinterpret_cast<const bf16*>(&tk);
      const bf16* ev = reinterpret_cast<const bf16*>(&tv);
      int s = s8 * 8;
#pragma unroll
      for (int u = 0; u < 8; ++u) {
        ks[c * HD + s + u] = (bf16)((float)ek[u] * rn);
        vt[(s + u) * CHK + c] = ev[u];
      }
    }
    beta_s[c] = betaw[(tok0 + c) * NH + h];
    lab[c] = law[(tok0 + c) * NH + h];
    rowsq[c] = 0.f;
  }
  wait_asynccnt0();
  __syncthreads();
  if (tid == 0) {
    float run = 0.f;
    for (int c = 0; c < CHK; ++c) { run += lab[c]; cla[c] = run; }
  }
  __syncthreads();
  const float claEnd = cla[CHK - 1];
  for (int i = tid; i < CHK * CHK; i += 256) {
    int c = i >> 7, e = i & 127;
    float sc = beta_s[c] * expf(claEnd - cla[c]);
    kwt[e * CHK + c] = (bf16)((float)ks[c * HD + e] * sc);
  }
  __syncthreads();

  v8f acc[2][4];
  // ---- matmul 1: attn = q @ k^T (contraction over head dim s)
#pragma unroll
  for (int i = 0; i < 2; ++i)
#pragma unroll
    for (int j = 0; j < 4; ++j) acc[i][j] = {};
#pragma unroll
  for (int k0 = 0; k0 < HD; k0 += 32) {
    v16bf a0 = load_frag_a(qs, HD, wm, k0, lane);
    v16bf a1 = load_frag_a(qs, HD, wm + 16, k0, lane);
#pragma unroll
    for (int ni = 0; ni < 4; ++ni) {
      v16bf bfr = load_frag_b(ks, HD, wn + ni * 16, k0, lane);
      acc[0][ni] = wmma_bf16(a0, bfr, acc[0][ni]);
      acc[1][ni] = wmma_bf16(a1, bfr, acc[1][ni]);
    }
  }
  __syncthreads();  // all reads of qs finished; reuse qs for attn
#pragma unroll
  for (int mi = 0; mi < 2; ++mi)
#pragma unroll
    for (int ni = 0; ni < 4; ++ni)
#pragma unroll
      for (int v = 0; v < 8; ++v) {
        int r, cc;
        cd_coord(lane, v, r, cc);
        int cg = wm + mi * 16 + r;
        int eg = wn + ni * 16 + cc;
        float val = acc[mi][ni][v] * beta_s[eg] * expf(cla[cg] - cla[eg]);
        if (eg > cg) val = 0.f;
        qs[cg * CHK + eg] = (bf16)val;
      }
  __syncthreads();

  // ---- matmul 2: y_local = attn @ v (contraction over e)
#pragma unroll
  for (int i = 0; i < 2; ++i)
#pragma unroll
    for (int j = 0; j < 4; ++j) acc[i][j] = {};
#pragma unroll
  for (int k0 = 0; k0 < CHK; k0 += 32) {
    v16bf a0 = load_frag_a(qs, CHK, wm, k0, lane);
    v16bf a1 = load_frag_a(qs, CHK, wm + 16, k0, lane);
#pragma unroll
    for (int ni = 0; ni < 4; ++ni) {
      v16bf bfr = load_frag_b(vt, CHK, wn + ni * 16, k0, lane);
      acc[0][ni] = wmma_bf16(a0, bfr, acc[0][ni]);
      acc[1][ni] = wmma_bf16(a1, bfr, acc[1][ni]);
    }
  }
#pragma unroll
  for (int mi = 0; mi < 2; ++mi)
#pragma unroll
    for (int ni = 0; ni < 4; ++ni)
#pragma unroll
      for (int v = 0; v < 8; ++v) {
        int r, cc;
        cd_coord(lane, v, r, cc);
        int cg = wm + mi * 16 + r;
        int dg = wn + ni * 16 + cc;
        float val = acc[mi][ni][v];
        ylocal[(bnh * CHK + cg) * HD + dg] = val;
        atomicAdd(&rowsq[cg], val * val);
      }

  // ---- matmul 3: delta = v^T @ kw (contraction over c)
#pragma unroll
  for (int i = 0; i < 2; ++i)
#pragma unroll
    for (int j = 0; j < 4; ++j) acc[i][j] = {};
#pragma unroll
  for (int k0 = 0; k0 < CHK; k0 += 32) {
    v16bf a0 = load_frag_a(vt, CHK, wm, k0, lane);
    v16bf a1 = load_frag_a(vt, CHK, wm + 16, k0, lane);
#pragma unroll
    for (int ni = 0; ni < 4; ++ni) {
      v16bf bfr = load_frag_b(kwt, CHK, wn + ni * 16, k0, lane);
      acc[0][ni] = wmma_bf16(a0, bfr, acc[0][ni]);
      acc[1][ni] = wmma_bf16(a1, bfr, acc[1][ni]);
    }
  }
#pragma unroll
  for (int mi = 0; mi < 2; ++mi)
#pragma unroll
    for (int ni = 0; ni < 4; ++ni)
#pragma unroll
      for (int v = 0; v < 8; ++v) {
        int r, cc;
        cd_coord(lane, v, r, cc);
        int dg = wm + mi * 16 + r;
        int eg = wn + ni * 16 + cc;
        deltaw[(bnh * HD + dg) * HD + eg] = acc[mi][ni][v];
      }
  __syncthreads();
  if (tid < CHK) cdw[bnh * CHK + tid] = expf(cla[tid]);
  if (tid == 0) {
    tdw[bnh] = expf(claEnd);
    float s = 0.f;
    for (int c = 0; c < CHK; ++c) s += sqrtf(rowsq[c]);
    atomicAdd(&sacc[b], s);
  }
}

// ---------------------------------------------------------------- scan
__global__ __launch_bounds__(256) void scan_chunks(
    const bf16* __restrict__ qb, const float* __restrict__ ylocal,
    const float* __restrict__ deltaw, const float* __restrict__ cdw,
    const float* __restrict__ tdw, float* __restrict__ yout,
    float* __restrict__ fsout) {
  __shared__ float st[HD * HD];               // state[d][e] f32
  __shared__ alignas(16) bf16 stb[HD * HD];   // bf16 copy, [d][e]
  __shared__ alignas(16) bf16 qs[CHK * HD];
  const int h = blockIdx.x, b = blockIdx.y;
  const int tid = threadIdx.x, lane = tid & 31, wave = tid >> 5;
  const int wm = (wave >> 1) * 32, wn = (wave & 1) * 64;

  for (int i = tid; i < HD * HD; i += 256) st[i] = 0.f;
  __syncthreads();

  for (int n = 0; n < NCH; ++n) {
    const size_t tok0 = (size_t)b * TT + (size_t)n * CHK;
    const size_t bnh = (size_t)(b * NCH + n) * NH + h;
    // q chunk via async DMA; overlap with state->bf16 conversion
    for (int i = tid; i < (CHK * HD) / 8; i += 256) {
      int c = i >> 4, part = i & 15;
      async_load_lds_b128(&qs[c * HD + part * 8],
                          &qb[(tok0 + c) * DIM + (size_t)h * HD + part * 8]);
    }
    for (int i = tid; i < HD * HD; i += 256) stb[i] = (bf16)st[i];
    wait_asynccnt0();
    __syncthreads();

    v8f acc[2][4];
#pragma unroll
    for (int i = 0; i < 2; ++i)
#pragma unroll
      for (int j = 0; j < 4; ++j) acc[i][j] = {};
#pragma unroll
    for (int k0 = 0; k0 < HD; k0 += 32) {
      v16bf a0 = load_frag_a(qs, HD, wm, k0, lane);
      v16bf a1 = load_frag_a(qs, HD, wm + 16, k0, lane);
#pragma unroll
      for (int ni = 0; ni < 4; ++ni) {
        v16bf bfr = load_frag_b(stb, HD, wn + ni * 16, k0, lane);
        acc[0][ni] = wmma_bf16(a0, bfr, acc[0][ni]);
        acc[1][ni] = wmma_bf16(a1, bfr, acc[1][ni]);
      }
    }
#pragma unroll
    for (int mi = 0; mi < 2; ++mi)
#pragma unroll
      for (int ni = 0; ni < 4; ++ni)
#pragma unroll
        for (int v = 0; v < 8; ++v) {
          int r, cc;
          cd_coord(lane, v, r, cc);
          int cg = wm + mi * 16 + r;
          int dg = wn + ni * 16 + cc;
          float cdv = cdw[bnh * CHK + cg];
          float yl = ylocal[(bnh * CHK + cg) * HD + dg];
          yout[(tok0 + cg) * DIM + (size_t)h * HD + dg] =
              acc[mi][ni][v] * cdv + yl;
        }
    // state update (matmul used stb copy, so no hazard with st writes)
    float td = tdw[bnh];
    for (int i = tid; i < HD * HD; i += 256)
      st[i] = st[i] * td + deltaw[bnh * (size_t)(HD * HD) + i];
    __syncthreads();
  }
  for (int i = tid; i < HD * HD; i += 256)
    fsout[(size_t)(b * NH + h) * HD * HD + i] = st[i];
}

// ---------------------------------------------------------------- tiny utils
__global__ void zero_acc(float* sacc) {
  if (threadIdx.x < 2) sacc[threadIdx.x] = 0.f;
}
__global__ void finalize_surprise(const float* sacc, float* sur_out) {
  if (threadIdx.x < 2)
    sur_out[threadIdx.x] = sacc[threadIdx.x] * (1.f / (NCH * NH * CHK));
}

// ---------------------------------------------------------------- launcher
extern "C" void kernel_launch(void* const* d_in, const int* in_sizes, int n_in,
                              void* d_out, int out_size, void* d_ws,
                              size_t ws_size, hipStream_t stream) {
  (void)in_sizes; (void)n_in; (void)out_size; (void)ws_size;
  const float* x      = (const float*)d_in[0];
  const float* Wq     = (const float*)d_in[1];
  const float* bq     = (const float*)d_in[2];
  const float* Wk     = (const float*)d_in[3];
  const float* bk     = (const float*)d_in[4];
  const float* Wv     = (const float*)d_in[5];
  const float* bv     = (const float*)d_in[6];
  const float* Wbeta  = (const float*)d_in[7];
  const float* bbeta  = (const float*)d_in[8];
  const float* Walpha = (const float*)d_in[9];
  const float* balpha = (const float*)d_in[10];

  float* out = (float*)d_out;
  float* y_out = out;                                  // B*T*DIM
  float* fs_out = out + (size_t)BB * TT * DIM;         // B*NH*HD*HD
  float* sur_out = fs_out + (size_t)BB * NH * HD * HD; // 2
  float* alpha_out = sur_out + 2;                      // B*T*NH

  char* ws = (char*)d_ws;
  size_t off = 0;
  bf16* xb  = (bf16*)(ws + off); off += (size_t)BT * DIM * sizeof(bf16);
  bf16* wbt = (bf16*)(ws + off); off += (size_t)3 * DIM * DIM * sizeof(bf16);
  bf16* qb  = (bf16*)(ws + off); off += (size_t)BT * DIM * sizeof(bf16);
  bf16* kb  = (bf16*)(ws + off); off += (size_t)BT * DIM * sizeof(bf16);
  bf16* vb  = (bf16*)(ws + off); off += (size_t)BT * DIM * sizeof(bf16);
  float* betaw = (float*)(ws + off); off += (size_t)BT * NH * sizeof(float);
  float* law   = (float*)(ws + off); off += (size_t)BT * NH * sizeof(float);
  float* ylocal = (float*)(ws + off);
  off += (size_t)BB * NCH * NH * CHK * HD * sizeof(float);
  float* deltaw = (float*)(ws + off);
  off += (size_t)BB * NCH * NH * HD * HD * sizeof(float);
  float* cdw = (float*)(ws + off);
  off += (size_t)BB * NCH * NH * CHK * sizeof(float);
  float* tdw = (float*)(ws + off);
  off += (size_t)BB * NCH * NH * sizeof(float);
  float* sacc = (float*)(ws + off);

  // 1. conversions: x -> bf16; W -> transposed bf16 (enables async B staging)
  cvt_f32_bf16<<<4096, 256, 0, stream>>>(x, xb, (long long)BT * DIM);
  cvt_transpose_w<<<dim3(DIM / 32, DIM / 32), dim3(32, 8), 0, stream>>>(
      Wq, wbt);
  cvt_transpose_w<<<dim3(DIM / 32, DIM / 32), dim3(32, 8), 0, stream>>>(
      Wk, wbt + (size_t)DIM * DIM);
  cvt_transpose_w<<<dim3(DIM / 32, DIM / 32), dim3(32, 8), 0, stream>>>(
      Wv, wbt + (size_t)2 * DIM * DIM);
  // 2. q/k/v projections (WMMA GEMM, both tiles async-DMA, bf16 out)
  gemm_qkv<<<dim3(DIM / 128, BT / 128, 3), 256, 0, stream>>>(
      xb, wbt, bq, bk, bv, qb, kb, vb);
  // 3. gates
  proj_gate<<<BT, 256, 0, stream>>>(x, Wbeta, bbeta, Walpha, balpha, betaw,
                                    law, alpha_out);
  // 4. chunk-local (parallel over b,n,h)
  zero_acc<<<1, 32, 0, stream>>>(sacc);
  chunk_local<<<dim3(NH, NCH, BB), 256, 0, stream>>>(
      qb, kb, vb, betaw, law, ylocal, deltaw, cdw, tdw, sacc);
  // 5. sequential inter-chunk scan (parallel over b,h)
  scan_chunks<<<dim3(NH, BB), 256, 0, stream>>>(qb, ylocal, deltaw, cdw, tdw,
                                                y_out, fs_out);
  // 6. surprise
  finalize_surprise<<<1, 32, 0, stream>>>(sacc, sur_out);
}
